// ResidualGATBlock_20899310863066
// MI455X (gfx1250) — matrix operand order repaired and checked by reading, not compiled, not run on hardware
//
#include <hip/hip_runtime.h>
#include <hip/hip_bf16.h>

// Problem constants from the reference: D = H*C = 128, H = 4, C = 32.
// N, E derived from in_sizes. N (=50000) is divisible by 16.

typedef __attribute__((ext_vector_type(16))) __bf16 v16bf;
typedef __attribute__((ext_vector_type(8)))  float  v8f;

// ---- float <-> order-preserving uint (for atomic max on floats) ----
__device__ __forceinline__ unsigned encf(float f) {
    unsigned u = __float_as_uint(f);
    return (u & 0x80000000u) ? ~u : (u | 0x80000000u);
}
__device__ __forceinline__ float decf(unsigned e) {
    unsigned u = (e & 0x80000000u) ? (e ^ 0x80000000u) : ~e;
    return __uint_as_float(u);
}
#define ENC_NEG_INF 0x007FFFFFu   // encf(-inf)

// -------------------- init: zero accumulators, set max = -inf --------------------
__global__ __launch_bounds__(256) void init_kernel(float* __restrict__ agg,
                                                   float* __restrict__ denom,
                                                   float* __restrict__ a_s,
                                                   float* __restrict__ a_d,
                                                   unsigned* __restrict__ mmax,
                                                   int N) {
    int t = blockIdx.x * blockDim.x + threadIdx.x;
    int tot = N * 128;
    if (t < tot) agg[t] = 0.0f;
    int t4 = N * 4;
    if (t < t4) {
        denom[t] = 0.0f;
        a_s[t]   = 0.0f;
        a_d[t]   = 0.0f;
        mmax[t]  = ENC_NEG_INF;
    }
}

// ---------- fused LayerNorm + dual GEMM (WMMA bf16) + attention dots ----------
// Block = 256 threads = 8 waves; block owns 16 node rows.
// Wave w computes output columns [16w, 16w+16) for BOTH GEMMs (K = 128 via 4 WMMAs).
__global__ __launch_bounds__(256) void ln_dualgemm_kernel(
    const float* __restrict__ x,
    const float* __restrict__ gamma, const float* __restrict__ beta,
    const float* __restrict__ W_gat, const float* __restrict__ W_proj,
    const float* __restrict__ b_proj,
    const float* __restrict__ att_src, const float* __restrict__ att_dst,
    float* __restrict__ identity, float* __restrict__ xw,
    float* __restrict__ a_s, float* __restrict__ a_d, int N)
{
    __shared__ __bf16 sIdent[16][128];  // x in bf16        (A for W_proj GEMM)
    __shared__ __bf16 sXn[16][128];     // LayerNorm(x) bf16 (A for W_gat GEMM)

    const int tid = threadIdx.x;
    const int nodeBase = blockIdx.x * 16;

    // ---- load 16x128 tile + LayerNorm (16 threads per row, 8 elems each) ----
    {
        const int row = tid >> 4;       // 0..15
        const int l16 = tid & 15;
        int node = nodeBase + row;
        if (node >= N) node = N - 1;    // clamp (N % 16 == 0 in practice)
        float v[8];
        float s = 0.0f, ss = 0.0f;
        #pragma unroll
        for (int k = 0; k < 8; ++k) {
            float f = x[(size_t)node * 128 + l16 + 16 * k];
            v[k] = f; s += f; ss += f * f;
        }
        #pragma unroll
        for (int m = 1; m < 16; m <<= 1) {  // 16-lane group reduce (one row per half-wave)
            s  += __shfl_xor(s,  m);
            ss += __shfl_xor(ss, m);
        }
        const float mu   = s * (1.0f / 128.0f);
        const float var  = ss * (1.0f / 128.0f) - mu * mu;
        const float rstd = rsqrtf(var + 1e-5f);
        #pragma unroll
        for (int k = 0; k < 8; ++k) {
            int j = l16 + 16 * k;
            sIdent[row][j] = (__bf16)v[k];
            sXn[row][j]    = (__bf16)((v[k] - mu) * rstd * gamma[j] + beta[j]);
        }
    }
    __syncthreads();

    // ---- WMMA: each wave does a 16(M) x 16(N) tile, K=128 in 4 steps of 32 ----
    const int wv      = tid >> 5;       // wave 0..7 -> column tile
    const int lane    = tid & 31;
    const int colBase = wv * 16;
    const int nlo     = lane & 15;
    const int g16     = lane >> 4;      // which half-wave (K-half selector)
    const int col     = colBase + nlo;

    for (int g = 0; g < 2; ++g) {
        const __bf16 (*A)[128] = g ? sXn : sIdent;
        const float* W = g ? W_gat : W_proj;

        v8f acc = {};
        #pragma unroll
        for (int kk = 0; kk < 4; ++kk) {
            const int kb = kk * 32;
            v16bf a, b;
            // A 16x32 bf16 layout: lane row = lane%16; elems 0..7 -> K = kb + g16*8 + e,
            // elems 8..15 -> K = kb + 16 + g16*8 + e.
            #pragma unroll
            for (int e = 0; e < 8; ++e) {
                a[e]     = A[nlo][kb + g16 * 8 + e];
                a[e + 8] = A[nlo][kb + 16 + g16 * 8 + e];
            }
            // B 32x16 bf16 layout: lane col = lane%16; elems 0..7 -> K = kb + g16*8 + e,
            // elems 8..15 -> K = kb + 16 + g16*8 + (e-8).
            #pragma unroll
            for (int e = 0; e < 16; ++e) {
                b[e] = (__bf16)W[(size_t)(kb + g16 * 8 + (e & 7) + ((e >> 3) << 4)) * 128 + col];
            }
            acc = __builtin_amdgcn_wmma_f32_16x16x32_bf16(
                false, a, false, b, (short)0, acc, false, false);
        }

        // D layout: lane holds column `col`; vgpr j -> row m = g16*8 + j.
        if (g == 0) {
            const float bb = b_proj[col];
            #pragma unroll
            for (int j = 0; j < 8; ++j) {
                int m = g16 * 8 + j;
                int node = nodeBase + m;
                if (node < N)
                    identity[(size_t)node * 128 + col] = acc[j] + bb;
            }
        } else {
            const int h = colBase >> 5;                 // head of this column tile
            const float asv = att_src[(h << 5) + (col & 31)];
            const float adv = att_dst[(h << 5) + (col & 31)];
            #pragma unroll
            for (int j = 0; j < 8; ++j) {
                int m = g16 * 8 + j;
                int node = nodeBase + m;
                if (node < N)
                    xw[(size_t)node * 128 + col] = acc[j];
            }
            // per-row 16-lane reductions for a_s / a_d, two waves per head each add half
            #pragma unroll
            for (int j = 0; j < 8; ++j) {
                float ps = acc[j] * asv;
                float pd = acc[j] * adv;
                #pragma unroll
                for (int m2 = 1; m2 < 16; m2 <<= 1) {
                    ps += __shfl_xor(ps, m2);
                    pd += __shfl_xor(pd, m2);
                }
                if (nlo == 0) {
                    int node = nodeBase + g16 * 8 + j;
                    if (node < N) {
                        atomicAdd(&a_s[node * 4 + h], ps);
                        atomicAdd(&a_d[node * 4 + h], pd);
                    }
                }
            }
        }
    }
}

// -------------------- edge pass 1: segment max over destinations --------------------
__global__ __launch_bounds__(256) void edge_max_kernel(
    const int* __restrict__ ei, int E,
    const float* __restrict__ a_s, const float* __restrict__ a_d,
    unsigned* __restrict__ mmax, int total)
{
    int t = blockIdx.x * blockDim.x + threadIdx.x;
    if (t >= total) return;
    int s, d;
    if (t < E) { s = ei[t]; d = ei[E + t]; }
    else       { s = d = t - E; }                    // self loops
    float4 as4 = ((const float4*)a_s)[s];
    float4 ad4 = ((const float4*)a_d)[d];
    float e0 = as4.x + ad4.x, e1 = as4.y + ad4.y;
    float e2 = as4.z + ad4.z, e3 = as4.w + ad4.w;
    e0 = e0 > 0.0f ? e0 : 0.2f * e0;
    e1 = e1 > 0.0f ? e1 : 0.2f * e1;
    e2 = e2 > 0.0f ? e2 : 0.2f * e2;
    e3 = e3 > 0.0f ? e3 : 0.2f * e3;
    atomicMax(&mmax[d * 4 + 0], encf(e0));
    atomicMax(&mmax[d * 4 + 1], encf(e1));
    atomicMax(&mmax[d * 4 + 2], encf(e2));
    atomicMax(&mmax[d * 4 + 3], encf(e3));
}

// ------------- edge pass 2: exp weights, denom, unnormalized aggregation -------------
// One wave32 per edge: lane L loads float4 of xw[src] (channels 4L..4L+3, head = L>>3).
__global__ __launch_bounds__(256) void edge_agg_kernel(
    const int* __restrict__ ei, int E, int total,
    const float* __restrict__ a_s, const float* __restrict__ a_d,
    const unsigned* __restrict__ mmax,
    const float* __restrict__ xw,
    float* __restrict__ denom, float* __restrict__ agg)
{
    const int lane = threadIdx.x & 31;
    const int e = blockIdx.x * 8 + (threadIdx.x >> 5);
    if (e >= total) return;
    int s, d;
    if (e < E) { s = ei[e]; d = ei[E + e]; }
    else       { s = d = e - E; }

    float w = 0.0f;
    if (lane < 4) {                                   // one head per lane
        float ev = a_s[s * 4 + lane] + a_d[d * 4 + lane];
        ev = ev > 0.0f ? ev : 0.2f * ev;
        float m = decf(mmax[d * 4 + lane]);
        w = __expf(ev - m);
        atomicAdd(&denom[d * 4 + lane], w);
    }
    const float wh = __shfl(w, lane >> 3);            // broadcast this lane's head weight

    float4 xv = ((const float4*)xw)[(size_t)s * 32 + lane];
    float* ap = &agg[(size_t)d * 128 + lane * 4];
    atomicAdd(ap + 0, wh * xv.x);
    atomicAdd(ap + 1, wh * xv.y);
    atomicAdd(ap + 2, wh * xv.z);
    atomicAdd(ap + 3, wh * xv.w);
}

// -------------------- finalize: relu(identity + agg/denom + bias) --------------------
__global__ __launch_bounds__(256) void finalize_kernel(
    const float* __restrict__ identity, const float* __restrict__ agg,
    const float* __restrict__ denom, const float* __restrict__ bias_gat,
    float* __restrict__ out, int N)
{
    int t = blockIdx.x * blockDim.x + threadIdx.x;
    if (t >= N * 128) return;
    int col = t & 127;
    int node = t >> 7;
    int h = col >> 5;
    float o = identity[t] + agg[t] / (denom[node * 4 + h] + 1e-16f) + bias_gat[col];
    out[t] = o > 0.0f ? o : 0.0f;
}

extern "C" void kernel_launch(void* const* d_in, const int* in_sizes, int n_in,
                              void* d_out, int out_size, void* d_ws, size_t ws_size,
                              hipStream_t stream)
{
    const float* x        = (const float*)d_in[0];
    const int*   ei       = (const int*)  d_in[1];
    const float* gamma    = (const float*)d_in[2];
    const float* beta     = (const float*)d_in[3];
    const float* W_gat    = (const float*)d_in[4];
    const float* att_src  = (const float*)d_in[5];
    const float* att_dst  = (const float*)d_in[6];
    const float* bias_gat = (const float*)d_in[7];
    const float* W_proj   = (const float*)d_in[8];
    const float* b_proj   = (const float*)d_in[9];

    const int N = in_sizes[0] / 128;
    const int E = in_sizes[1] / 2;
    const int total = E + N;

    // workspace layout (floats): identity | xw | agg | a_s | a_d | denom | mmax(u32)
    float* wsf = (float*)d_ws;
    size_t n = (size_t)N;
    float*    identity = wsf;
    float*    xw       = wsf + 128 * n;
    float*    agg      = wsf + 256 * n;
    float*    a_s      = wsf + 384 * n;
    float*    a_d      = a_s + 4 * n;
    float*    denom    = a_d + 4 * n;
    unsigned* mmax     = (unsigned*)(denom + 4 * n);

    (void)n_in; (void)out_size; (void)ws_size;

    {
        int tot = N * 128;
        init_kernel<<<(tot + 255) / 256, 256, 0, stream>>>(agg, denom, a_s, a_d, mmax, N);
    }
    ln_dualgemm_kernel<<<(N + 15) / 16, 256, 0, stream>>>(
        x, gamma, beta, W_gat, W_proj, b_proj, att_src, att_dst,
        identity, xw, a_s, a_d, N);
    edge_max_kernel<<<(total + 255) / 256, 256, 0, stream>>>(ei, E, a_s, a_d, mmax, total);
    edge_agg_kernel<<<(total + 7) / 8, 256, 0, stream>>>(
        ei, E, total, a_s, a_d, mmax, xw, denom, agg);
    finalize_kernel<<<(N * 128 + 255) / 256, 256, 0, stream>>>(
        identity, agg, denom, bias_gat, (float*)d_out, N);
}
